// LatentODE_16243566313584
// MI455X (gfx1250) — compile-verified, matrix operand order checked by reading, not run on hardware
//
#include <hip/hip_runtime.h>
#include <hip/hip_bf16.h>
#include <stdint.h>
#include <stddef.h>

// ---------------------------------------------------------------------------
// Latent ODE on MI455X (gfx1250): bf16 WMMA (f32 accum) persistent kernels.
// B=512,T=200,D=64,H=256,L=64,W=256.  Compute-bound (~290 GF vs ~30 MB moved).
// One workgroup (8 wave32 waves) per 16 batch rows; the two serial scans
// (GRU over T, RK ODE over T-1 x 6 stages) run inside persistent kernels.
// Weights are packed once into CDNA5 WMMA fragment layout (ISA 7.12.2) and
// stay resident in the 192MB L2.  Per-element GEMM epilogues are fused
// in-register on the C fragment and scattered straight into the next layer's
// packed-A LDS buffer (one barrier per GEMM).  Each wave processes its two
// N-tiles jointly (one ds_load A chain feeds two v_wmma streams), with outer
// tile loops kept ROLLED to avoid VGPR spills.
// ---------------------------------------------------------------------------

typedef __attribute__((ext_vector_type(16))) __bf16 v16bf;
typedef __attribute__((ext_vector_type(8)))  float  v8f;

#define BB 512
#define TT 200
#define DD 64
#define HH 256
#define LL 64
#define G3 768
#define DT_F 0.4f

__device__ __constant__ float c_ANEXT[5][5] = {
  { 0.161f, 0.f, 0.f, 0.f, 0.f },
  { -0.008480655492356989f, 0.335480655492357f, 0.f, 0.f, 0.f },
  { 2.8971530571054935f, -6.359448489975075f, 4.3622954328695815f, 0.f, 0.f },
  { 5.325864828439257f, -11.748883564062828f, 7.4955393428898365f, -0.09249506636175525f, 0.f },
  { 5.86145544294642f, -12.92096931784711f, 8.159367898576159f, -0.071584973281401f, -0.028269050394068383f },
};
__device__ __constant__ float c_BW[6] = {
  0.09646076681806523f, 0.01f, 0.4798896504144996f,
  1.379008574103742f, -3.290069515436081f, 2.324710524099774f };

// CDNA5 WMMA 16-bit A/B fragment packing (ISA 7.12.2):
//   lane = row + 16*g, halfword slot = 2*j + h, where for k<16: g=k>>3,
//   j=(k&7)>>1; for k>=16: g=(k-16)>>3, j=4+(((k-16)&7)>>1); h=k&1.
__device__ __forceinline__ int pack_idx(int row, int k) {
  int h = k & 1, g, j;
  if (k < 16) { g = k >> 3; j = (k & 7) >> 1; }
  else        { int k2 = k - 16; g = k2 >> 3; j = 4 + ((k2 & 7) >> 1); }
  return (row + 16 * g) * 16 + 2 * j + h;
}

__device__ __forceinline__ float sigmoidf_(float x) { return 1.f / (1.f + __expf(-x)); }
__device__ __forceinline__ float softplusf_(float x) {
  if (x > 20.f) return x;
  if (x < -20.f) return __expf(x);
  return log1pf(__expf(x));
}

// Single-tile K-loop: A from LDS, B from global (L2-resident packed weights)
__device__ __forceinline__ v8f wmma_kloop(const __bf16* aLDS, const __bf16* bGLB,
                                          int nkb, int lane) {
  v8f acc = {0.f,0.f,0.f,0.f,0.f,0.f,0.f,0.f};
  for (int kb = 0; kb < nkb; ++kb) {
    if (kb + 1 < nkb) __builtin_prefetch((const void*)(bGLB + (size_t)(kb + 1) * 512), 0, 3);
    v16bf a = *(const v16bf*)(aLDS + (size_t)kb * 512 + lane * 16);
    v16bf b = *(const v16bf*)(bGLB + (size_t)kb * 512 + lane * 16);
    acc = __builtin_amdgcn_wmma_f32_16x16x32_bf16(false, a, false, b, (short)0, acc,
                                                  false, false);
  }
  return acc;
}

// Dual-tile K-loop: one A fragment load feeds two B streams / two v_wmma.
// Accumulates on top of acc0/acc1 (callers pass zero or partial sums).
__device__ __forceinline__ void wmma_kloop2(const __bf16* aLDS, const __bf16* b0g,
                                            const __bf16* b1g, int nkb, int lane,
                                            v8f& acc0, v8f& acc1) {
  for (int kb = 0; kb < nkb; ++kb) {
    if (kb + 1 < nkb) {
      __builtin_prefetch((const void*)(b0g + (size_t)(kb + 1) * 512), 0, 3);
      __builtin_prefetch((const void*)(b1g + (size_t)(kb + 1) * 512), 0, 3);
    }
    v16bf a  = *(const v16bf*)(aLDS + (size_t)kb * 512 + lane * 16);
    v16bf b0 = *(const v16bf*)(b0g  + (size_t)kb * 512 + lane * 16);
    v16bf b1 = *(const v16bf*)(b1g  + (size_t)kb * 512 + lane * 16);
    acc0 = __builtin_amdgcn_wmma_f32_16x16x32_bf16(false, a, false, b0, (short)0, acc0,
                                                   false, false);
    acc1 = __builtin_amdgcn_wmma_f32_16x16x32_bf16(false, a, false, b1, (short)0, acc1,
                                                   false, false);
  }
}

// Scatter a 16x16 f32 C fragment to LDS staging (C layout: lane=n+16g, vgpr j -> m=j+8g)
__device__ __forceinline__ void store_tile(float* stage, int ld, int colBase,
                                           v8f acc, int lane) {
  int n = lane & 15, g = lane >> 4;
  float* ap = (float*)&acc;
#pragma unroll
  for (int j = 0; j < 8; ++j) stage[(j + 8 * g) * ld + colBase + n] = ap[j];
}

// Fused per-element epilogue: act(acc + bias[k]) -> packed-A LDS. 0=relu,1=softplus
template <int ACT>
__device__ __forceinline__ void epi_store(v8f acc, int nt, const float* __restrict__ bias,
                                          __bf16* outPack, int n, int g) {
  int k = nt * 16 + n;
  float bk = bias[k];
  int blk = (k >> 5) * 512, kk = k & 31;
  float* ap = (float*)&acc;
#pragma unroll
  for (int j = 0; j < 8; ++j) {
    int m = j + 8 * g;
    float v = ap[j] + bk;
    v = (ACT == 0) ? (v > 0.f ? v : 0.f) : softplusf_(v);
    outPack[blk + pack_idx(m, kk)] = (__bf16)v;
  }
}

// 16-row x 256-col GEMM (16 tiles over 8 waves = 1 pair/wave) + fused act
template <int ACT>
__device__ __forceinline__ void gemm16_fused_act(const __bf16* aLDS, const __bf16* w,
                                                 const float* __restrict__ bias,
                                                 __bf16* outPack, int nkb) {
  int tid = threadIdx.x, lane = tid & 31, wave = tid >> 5;
  int n = lane & 15, g = lane >> 4;
  int nt0 = wave, nt1 = wave + 8;
  v8f acc0 = {0.f,0.f,0.f,0.f,0.f,0.f,0.f,0.f};
  v8f acc1 = acc0;
  wmma_kloop2(aLDS, w + (size_t)nt0 * nkb * 512, w + (size_t)nt1 * nkb * 512,
              nkb, lane, acc0, acc1);
  epi_store<ACT>(acc0, nt0, bias, outPack, n, g);
  epi_store<ACT>(acc1, nt1, bias, outPack, n, g);
}

// --------------------------- weight packing --------------------------------
__global__ void pack_weight_kernel(const float* __restrict__ W, __bf16* __restrict__ dst,
                                   int N, int Ksrc, int k0, int K) {
  int total = N * K;
  for (int i = blockIdx.x * blockDim.x + threadIdx.x; i < total;
       i += gridDim.x * blockDim.x) {
    int n = i / K, k = i % K;
    int nt = n >> 4, ni = n & 15, kb = k >> 5, kk = k & 31;
    dst[(size_t)(nt * (K >> 5) + kb) * 512 + pack_idx(ni, kk)] =
        (__bf16)W[(size_t)n * Ksrc + k0 + k];
  }
}

// --------------------------- GRU encoder -----------------------------------
__global__ __launch_bounds__(256) void gru_encode_kernel(
    const float* __restrict__ ts, const float* __restrict__ ys,
    const float* __restrict__ gru_wih, const float* __restrict__ gru_b,
    const float* __restrict__ gru_bn, const __bf16* __restrict__ pWih,
    const __bf16* __restrict__ pWhh, float* __restrict__ hidden) {
  extern __shared__ __align__(32) char smem[];
  float*  gS  = (float*)smem;                       // 16*768 (rz: ig+hg, n: ig)
  float*  hnS = gS + 16 * G3;                       // 16*256 (hn)
  float*  hS  = hnS + 16 * HH;                      // 16*256 current h
  float*  w0S = hS + 16 * HH;                       // 768 (wih col0, ts coeff)
  float*  gbS = w0S + G3;                           // 768
  float*  bnS = gbS + G3;                           // 256
  __bf16* pX  = (__bf16*)(bnS + HH);                // 2*512  packed x  (K=64)
  __bf16* pH  = pX + 2 * 512;                       // 8*512  packed h  (K=256)

  const int tid = threadIdx.x;
  const int lane = tid & 31;
  const int wave = tid >> 5;
  const int m0  = blockIdx.x * 16;

  for (int i = tid; i < G3; i += 256) { w0S[i] = gru_wih[(size_t)i * 65]; gbS[i] = gru_b[i]; }
  for (int i = tid; i < HH; i += 256) bnS[i] = gru_bn[i];
  for (int i = tid; i < 16 * HH; i += 256) hS[i] = 0.f;
  for (int i = tid; i < 8 * 512; i += 256) pH[i] = (__bf16)0.f;
  __syncthreads();

  for (int s = 0; s < TT; ++s) {
    const int t = TT - 1 - s;
    for (int i = tid; i < 16 * DD; i += 256) {
      int m = i >> 6, k = i & 63;
      float v = ys[((size_t)(m0 + m) * TT + t) * DD + k];
      pX[(k >> 5) * 512 + pack_idx(m, k & 31)] = (__bf16)v;
    }
    __syncthreads();
    // 48 gate tiles as 3 pairs/wave; pairs 0,1 are r/z (ig+hg accumulated in
    // the SAME registers), pair 2 is the n-gate (ig stored, then hg computed)
#pragma unroll 1
    for (int p = 0; p < 3; ++p) {
      int nt0 = wave + 16 * p, nt1 = nt0 + 8;
      v8f acc0 = {0.f,0.f,0.f,0.f,0.f,0.f,0.f,0.f};
      v8f acc1 = acc0;
      wmma_kloop2(pX, pWih + (size_t)nt0 * 2 * 512, pWih + (size_t)nt1 * 2 * 512,
                  2, lane, acc0, acc1);
      if (p < 2) {
        // continue accumulating h @ Whh^T on top of the ig partial sums
        wmma_kloop2(pH, pWhh + (size_t)nt0 * 8 * 512, pWhh + (size_t)nt1 * 8 * 512,
                    8, lane, acc0, acc1);
        store_tile(gS, G3, nt0 * 16, acc0, lane);
        store_tile(gS, G3, nt1 * 16, acc1, lane);
      } else {
        store_tile(gS, G3, nt0 * 16, acc0, lane);   // ig for n-gate
        store_tile(gS, G3, nt1 * 16, acc1, lane);
        v8f hg0 = {0.f,0.f,0.f,0.f,0.f,0.f,0.f,0.f};
        v8f hg1 = hg0;
        wmma_kloop2(pH, pWhh + (size_t)nt0 * 8 * 512, pWhh + (size_t)nt1 * 8 * 512,
                    8, lane, hg0, hg1);
        store_tile(hnS, HH, (nt0 - 32) * 16, hg0, lane);
        store_tile(hnS, HH, (nt1 - 32) * 16, hg1, lane);
      }
    }
    __syncthreads();
    // gate math; thread owns one hidden unit n across all 16 rows
    const int n = tid;
    const float tsv = ts[t];
    const float br = gbS[n] + tsv * w0S[n];
    const float bz = gbS[n + HH] + tsv * w0S[n + HH];
    const float bi2 = gbS[n + 2 * HH] + tsv * w0S[n + 2 * HH];
    const float bnv = bnS[n];
    for (int m = 0; m < 16; ++m) {
      float r  = sigmoidf_(gS[m * G3 + n] + br);
      float z  = sigmoidf_(gS[m * G3 + n + HH] + bz);
      float in = gS[m * G3 + n + 2 * HH] + bi2;
      float hn = hnS[m * HH + n];
      float nn = tanhf(in + r * (hn + bnv));
      float ho = hS[m * HH + n];
      float hv = (1.f - z) * nn + z * ho;
      hS[m * HH + n] = hv;
      pH[(n >> 5) * 512 + pack_idx(m, n & 31)] = (__bf16)hv;
    }
    __syncthreads();
  }
  for (int i = tid; i < 16 * HH; i += 256)
    hidden[(size_t)(m0 + (i >> 8)) * HH + (i & 255)] = hS[i];
}

// --------------------------- latent + decoder + ODE ------------------------
__global__ __launch_bounds__(256) void latent_ode_kernel(
    const float* __restrict__ hidden, const float* __restrict__ noise,
    const float* __restrict__ scale_p, const float* __restrict__ h2l_b,
    const __bf16* __restrict__ pH2L, const float* __restrict__ d_b0,
    const __bf16* __restrict__ pD0, const float* __restrict__ d_b1,
    const __bf16* __restrict__ pD1, const float* __restrict__ d_b2,
    const __bf16* __restrict__ pD2, const float* __restrict__ f_b0,
    const __bf16* __restrict__ pF0, const float* __restrict__ f_b1,
    const __bf16* __restrict__ pF1, const float* __restrict__ f_b2,
    const __bf16* __restrict__ pF2, const float* __restrict__ h2d_b,
    const __bf16* __restrict__ pH2D, const float* __restrict__ ys,
    float* __restrict__ out) {
  extern __shared__ __align__(32) char smem[];
  float*  kS    = (float*)smem;              // 5 * 16*256  (k1..k5)
  float*  yS    = kS + 5 * 16 * HH;          // 16*256  current y
  float*  ynS   = yS + 16 * HH;              // 16*256  y_next accumulator
  float*  stage = ynS + 16 * HH;             // 16*128  context staging only
  float*  redu  = stage + 16 * 128;          // [0..15] recon, [16..31] kl
  __bf16* pA    = (__bf16*)(redu + 32);      // 8*512 packed activations
  __bf16* pB    = pA + 8 * 512;              // 8*512 packed activations

  const int tid = threadIdx.x;
  const int lane = tid & 31;
  const int wave = tid >> 5;
  const int n    = lane & 15;
  const int g    = lane >> 4;
  const int m0   = blockIdx.x * 16;
  const float scl = scale_p[0];

  if (tid < 32) redu[tid] = 0.f;
  for (int i = tid; i < 16 * HH; i += 256) {
    int m = i >> 8, k = i & 255;
    pA[(k >> 5) * 512 + pack_idx(m, k & 31)] =
        (__bf16)hidden[(size_t)(m0 + m) * HH + k];
  }
  __syncthreads();
  {  // context (16 x 128): mean/logstd live in different tiles -> stage it
    for (int nt = wave; nt < 8; nt += 8) {
      v8f acc = wmma_kloop(pA, pH2L + (size_t)nt * 8 * 512, 8, lane);
      store_tile(stage, 128, nt * 16, acc, lane);
    }
  }
  __syncthreads();
  for (int i = tid; i < 16 * LL; i += 256) {
    int m = i >> 6, l = i & 63;
    float mean   = stage[m * 128 + l] + h2l_b[l];
    float logstd = stage[m * 128 + 64 + l] + h2l_b[64 + l];
    float stdv   = __expf(logstd);
    float lat    = mean + noise[(size_t)(m0 + m) * LL + l] * stdv;
    pB[(l >> 5) * 512 + pack_idx(m, l & 31)] = (__bf16)lat;
    atomicAdd(&redu[16 + m], 0.5f * (mean * mean + stdv * stdv - 2.f * logstd - 1.f));
  }
  __syncthreads();
  // decoder MLP: 64 -> 256 -> 256 -> 256 (relu, relu, identity)
  gemm16_fused_act<0>(pB, pD0, d_b0, pA, 2);
  __syncthreads();
  gemm16_fused_act<0>(pA, pD1, d_b1, pB, 8);
  __syncthreads();
  {  // final decoder layer: identity, also materialize yS
    int nt0 = wave, nt1 = wave + 8;
    v8f a0 = {0.f,0.f,0.f,0.f,0.f,0.f,0.f,0.f};
    v8f a1 = a0;
    wmma_kloop2(pB, pD2 + (size_t)nt0 * 8 * 512, pD2 + (size_t)nt1 * 8 * 512,
                8, lane, a0, a1);
#pragma unroll 1
    for (int h = 0; h < 2; ++h) {
      v8f acc = h ? a1 : a0;
      int nt = h ? nt1 : nt0;
      int k = nt * 16 + n;
      float bk = d_b2[k];
      int blk = (k >> 5) * 512, kk = k & 31;
      float* ap = (float*)&acc;
#pragma unroll
      for (int j = 0; j < 8; ++j) {
        int m = j + 8 * g;
        float v = ap[j] + bk;
        yS[m * HH + k] = v;
        pA[blk + pack_idx(m, kk)] = (__bf16)v;
      }
    }
  }
  __syncthreads();

  __bf16* bi = pA;
  __bf16* bo = pB;

  // reconstruction: t=0 fused projection epilogue (4 tiles, waves 0..3)
  for (int nt = wave; nt < 4; nt += 8) {
    v8f acc = wmma_kloop(bi, pH2D + (size_t)nt * 8 * 512, 8, lane);
    int d = nt * 16 + n;
    float bk = h2d_b[d];
    float* ap = (float*)&acc;
#pragma unroll
    for (int j = 0; j < 8; ++j) {
      int m = j + 8 * g;
      float e = ys[((size_t)(m0 + m) * TT + 0) * DD + d] - (ap[j] + bk);
      atomicAdd(&redu[m], e * e);
    }
  }
  __syncthreads();

  // ODE: 199 steps x 6 RK stages; each stage: 3 WMMA GEMMs, 3 barriers.
  // Stage loop kept rolled to bound code size / register pressure.
#pragma unroll 1
  for (int t = 1; t < TT; ++t) {
#pragma unroll 1
    for (int s = 1; s <= 6; ++s) {
      gemm16_fused_act<1>(bi, pF0, f_b0, bo, 8);
      __syncthreads();
      gemm16_fused_act<1>(bo, pF1, f_b1, bi, 8);
      __syncthreads();
      {  // layer 3 with fused RK-stage finalize epilogue (dual tile)
        int nt0 = wave, nt1 = wave + 8;
        v8f a0 = {0.f,0.f,0.f,0.f,0.f,0.f,0.f,0.f};
        v8f a1 = a0;
        wmma_kloop2(bi, pF2 + (size_t)nt0 * 8 * 512, pF2 + (size_t)nt1 * 8 * 512,
                    8, lane, a0, a1);
#pragma unroll 1
        for (int h = 0; h < 2; ++h) {
          v8f acc = h ? a1 : a0;
          int nt = h ? nt1 : nt0;
          int k = nt * 16 + n;
          float bk = f_b2[k];
          int blk = (k >> 5) * 512, kk = k & 31;
          float* ap = (float*)&acc;
#pragma unroll
          for (int j = 0; j < 8; ++j) {
            int m = j + 8 * g;
            int i = m * HH + k;
            float ksv = scl * tanhf(ap[j] + bk);
            float yv  = yS[i];
            float accv = (s == 1 ? yv : ynS[i]) + DT_F * c_BW[s - 1] * ksv;
            if (s < 6) {
              ynS[i] = accv;
              kS[(size_t)(s - 1) * 16 * HH + i] = ksv;
              float yt = yv + DT_F * c_ANEXT[s - 1][s - 1] * ksv;
              for (int q = 0; q < s - 1; ++q)
                yt += DT_F * c_ANEXT[s - 1][q] * kS[(size_t)q * 16 * HH + i];
              bo[blk + pack_idx(m, kk)] = (__bf16)yt;
            } else {
              yS[i] = accv;
              bo[blk + pack_idx(m, kk)] = (__bf16)accv;   // y_t packed
            }
          }
        }
      }
      __syncthreads();
      __bf16* tmp = bi; bi = bo; bo = tmp;   // 6 swaps -> bi back to pA
    }
    // fused projection + squared-error for this t
    for (int nt = wave; nt < 4; nt += 8) {
      v8f acc = wmma_kloop(bi, pH2D + (size_t)nt * 8 * 512, 8, lane);
      int d = nt * 16 + n;
      float bk = h2d_b[d];
      float* ap = (float*)&acc;
#pragma unroll
      for (int j = 0; j < 8; ++j) {
        int m = j + 8 * g;
        float e = ys[((size_t)(m0 + m) * TT + t) * DD + d] - (ap[j] + bk);
        atomicAdd(&redu[m], e * e);
      }
    }
    __syncthreads();
  }
  if (tid < 16) out[m0 + tid] = 0.5f * redu[tid] + redu[16 + tid];
}

// --------------------------- host launcher ---------------------------------
extern "C" void kernel_launch(void* const* d_in, const int* in_sizes, int n_in,
                              void* d_out, int out_size, void* d_ws, size_t ws_size,
                              hipStream_t stream) {
  (void)in_sizes; (void)n_in; (void)out_size; (void)ws_size;
  const float* ts      = (const float*)d_in[0];
  const float* ys      = (const float*)d_in[1];
  const float* noise   = (const float*)d_in[2];
  const float* scale   = (const float*)d_in[3];
  const float* f_w0    = (const float*)d_in[4];
  const float* f_b0    = (const float*)d_in[5];
  const float* f_w1    = (const float*)d_in[6];
  const float* f_b1    = (const float*)d_in[7];
  const float* f_w2    = (const float*)d_in[8];
  const float* f_b2    = (const float*)d_in[9];
  const float* gru_wih = (const float*)d_in[10];
  const float* gru_whh = (const float*)d_in[11];
  const float* gru_b   = (const float*)d_in[12];
  const float* gru_bn  = (const float*)d_in[13];
  const float* h2l_w   = (const float*)d_in[14];
  const float* h2l_b   = (const float*)d_in[15];
  const float* d_w0    = (const float*)d_in[16];
  const float* d_b0    = (const float*)d_in[17];
  const float* d_w1    = (const float*)d_in[18];
  const float* d_b1    = (const float*)d_in[19];
  const float* d_w2    = (const float*)d_in[20];
  const float* d_b2    = (const float*)d_in[21];
  const float* h2d_w   = (const float*)d_in[22];
  const float* h2d_b   = (const float*)d_in[23];
  float* out = (float*)d_out;

  char* ws = (char*)d_ws;
  size_t off = 0;
  auto carve = [&](size_t bytes) -> void* {
    void* p = ws + off;
    off += (bytes + 255) & ~(size_t)255;
    return p;
  };
  __bf16* pWih = (__bf16*)carve((size_t)G3 * 64  * 2);
  __bf16* pWhh = (__bf16*)carve((size_t)G3 * 256 * 2);
  __bf16* pH2L = (__bf16*)carve((size_t)128 * 256 * 2);
  __bf16* pD0  = (__bf16*)carve((size_t)256 * 64  * 2);
  __bf16* pD1  = (__bf16*)carve((size_t)256 * 256 * 2);
  __bf16* pD2  = (__bf16*)carve((size_t)256 * 256 * 2);
  __bf16* pF0  = (__bf16*)carve((size_t)256 * 256 * 2);
  __bf16* pF1  = (__bf16*)carve((size_t)256 * 256 * 2);
  __bf16* pF2  = (__bf16*)carve((size_t)256 * 256 * 2);
  __bf16* pH2D = (__bf16*)carve((size_t)64  * 256 * 2);
  float* hid   = (float*)carve((size_t)BB * HH * 4);

  dim3 pg(128), pb(256);
  pack_weight_kernel<<<pg, pb, 0, stream>>>(gru_wih, pWih, G3, 65, 1, 64);
  pack_weight_kernel<<<pg, pb, 0, stream>>>(gru_whh, pWhh, G3, 256, 0, 256);
  pack_weight_kernel<<<pg, pb, 0, stream>>>(h2l_w, pH2L, 128, 256, 0, 256);
  pack_weight_kernel<<<pg, pb, 0, stream>>>(d_w0, pD0, 256, 64, 0, 64);
  pack_weight_kernel<<<pg, pb, 0, stream>>>(d_w1, pD1, 256, 256, 0, 256);
  pack_weight_kernel<<<pg, pb, 0, stream>>>(d_w2, pD2, 256, 256, 0, 256);
  pack_weight_kernel<<<pg, pb, 0, stream>>>(f_w0, pF0, 256, 256, 0, 256);
  pack_weight_kernel<<<pg, pb, 0, stream>>>(f_w1, pF1, 256, 256, 0, 256);
  pack_weight_kernel<<<pg, pb, 0, stream>>>(f_w2, pF2, 256, 256, 0, 256);
  pack_weight_kernel<<<pg, pb, 0, stream>>>(h2d_w, pH2D, 64, 256, 0, 256);

  const size_t gruSmem = (size_t)(16 * G3 * 4) + 2 * (16 * HH * 4) + G3 * 4 * 2 +
                         HH * 4 + 2 * 512 * 2 + 8 * 512 * 2;
  gru_encode_kernel<<<dim3(BB / 16), dim3(256), gruSmem, stream>>>(
      ts, ys, gru_wih, gru_b, gru_bn, pWih, pWhh, hid);

  const size_t odeSmem = (size_t)(5 * 16 * HH * 4) + 2 * (16 * HH * 4) +
                         16 * 128 * 4 + 32 * 4 + 2 * (8 * 512 * 2);
  latent_ode_kernel<<<dim3(BB / 16), dim3(256), odeSmem, stream>>>(
      hid, noise, scale, h2l_b, pH2L, d_b0, pD0, d_b1, pD1, d_b2, pD2,
      f_b0, pF0, f_b1, pF1, f_b2, pF2, h2d_b, pH2D, ys, out);
}
